// Encoder_52252572123266
// MI455X (gfx1250) — compile-verified, matrix operand order checked by reading, not run on hardware
//
#include <hip/hip_runtime.h>
#include <stdint.h>
#include <stddef.h>

// Problem constants (from reference)
#define NN 20000
#define EE 320000
#define DN 128
#define HN 256

typedef __attribute__((ext_vector_type(16))) __bf16 v16bf;
typedef __attribute__((ext_vector_type(8)))  float  v8f;
typedef unsigned int v4u __attribute__((ext_vector_type(4)));
typedef float        v4f __attribute__((ext_vector_type(4)));

union Frag {
    v16bf v;
    v4u   q[2];
    unsigned short h[16];
};

__device__ __forceinline__ unsigned short f2bf(float f) {
    unsigned u = __builtin_bit_cast(unsigned, f);
    u += 0x7FFFu + ((u >> 16) & 1u);   // round-to-nearest-even
    return (unsigned short)(u >> 16);
}
__device__ __forceinline__ float bf2f(unsigned short h) {
    unsigned u = ((unsigned)h) << 16;
    return __builtin_bit_cast(float, u);
}

// ---------------- elementwise helpers ----------------

__global__ void k_cvt_bf16(const float* __restrict__ in,
                           unsigned short* __restrict__ out, int n) {
    int i = blockIdx.x * blockDim.x + threadIdx.x;
    if (i < n) out[i] = f2bf(in[i]);
}

// W is row-major (K, NC) f32; Wt is row-major (NC, K) bf16
__global__ void k_transpose_w(const float* __restrict__ W,
                              unsigned short* __restrict__ Wt, int K, int NC) {
    int i = blockIdx.x * blockDim.x + threadIdx.x;
    if (i >= K * NC) return;
    int k = i / NC, n = i - k * NC;
    Wt[n * K + k] = f2bf(W[i]);
}

__global__ void k_zero_f32(float* __restrict__ p, int n) {
    int i = blockIdx.x * blockDim.x + threadIdx.x;
    if (i < n) p[i] = 0.0f;
}

// ---------------- generic GEMM: C = relu(A @ W + b) ----------------
// A: (M, K) bf16 row-major.  Wt: (NC, K) bf16 row-major (pre-transposed W).
// Block = 256 threads = 8 waves; each wave owns a 16x64 output tile.
// A fragment is software-pipelined (next-step load issued before current
// consume); the four B fragments are loaded as a simultaneously-live batch
// so the scheduler can overlap loads with the 4 back-to-back WMMAs.
// __launch_bounds__(256, 2) relaxes the VGPR budget to avoid full
// s_wait_loadcnt 0x0 drains from register recycling.

template<int K, int NC, bool F32OUT>
__global__ __launch_bounds__(256, 2) void k_gemm_bias_relu(
    const unsigned short* __restrict__ A,
    const unsigned short* __restrict__ Wt,
    const float* __restrict__ bias,
    float* __restrict__ Cf,
    unsigned short* __restrict__ Cb,
    int M) {

    constexpr int WPM = NC / 64;   // waves per m-tile (4 for NC=256, 2 for NC=128)
    constexpr int MTB = 8 / WPM;   // m-tiles per block

    const int lane = threadIdx.x & 31;
    const int wave = threadIdx.x >> 5;
    const int l16  = lane & 15;
    const int half = lane >> 4;
    const int m0 = (blockIdx.x * MTB + wave / WPM) * 16;
    const int n0 = (wave % WPM) * 64;
    if (m0 >= M) return;

    const unsigned short* Ar = A  + (size_t)(m0 + l16) * K;
    const unsigned short* Br = Wt + (size_t)(n0 + l16) * K;

    v8f acc[4];
#pragma unroll
    for (int j = 0; j < 4; ++j) acc[j] = (v8f){0.f,0.f,0.f,0.f,0.f,0.f,0.f,0.f};

    // prologue: first A fragment
    Frag a;
    {
        const int kb = half * 8;
        a.q[0] = *(const v4u*)(Ar + kb);
        a.q[1] = *(const v4u*)(Ar + kb + 16);
    }

#pragma unroll
    for (int k0 = 0; k0 < K; k0 += 32) {
        const int kb = k0 + half * 8;
        // batch-load all four B fragments (simultaneously live)
        Frag b0, b1, b2, b3;
        b0.q[0] = *(const v4u*)(Br + kb);
        b0.q[1] = *(const v4u*)(Br + kb + 16);
        b1.q[0] = *(const v4u*)(Br + (size_t)16 * K + kb);
        b1.q[1] = *(const v4u*)(Br + (size_t)16 * K + kb + 16);
        b2.q[0] = *(const v4u*)(Br + (size_t)32 * K + kb);
        b2.q[1] = *(const v4u*)(Br + (size_t)32 * K + kb + 16);
        b3.q[0] = *(const v4u*)(Br + (size_t)48 * K + kb);
        b3.q[1] = *(const v4u*)(Br + (size_t)48 * K + kb + 16);
        // software-pipeline the next A fragment
        Frag an;
        if (k0 + 32 < K) {
            an.q[0] = *(const v4u*)(Ar + kb + 32);
            an.q[1] = *(const v4u*)(Ar + kb + 48);
        }
        acc[0] = __builtin_amdgcn_wmma_f32_16x16x32_bf16(
            false, a.v, false, b0.v, (short)0, acc[0], false, false);
        acc[1] = __builtin_amdgcn_wmma_f32_16x16x32_bf16(
            false, a.v, false, b1.v, (short)0, acc[1], false, false);
        acc[2] = __builtin_amdgcn_wmma_f32_16x16x32_bf16(
            false, a.v, false, b2.v, (short)0, acc[2], false, false);
        acc[3] = __builtin_amdgcn_wmma_f32_16x16x32_bf16(
            false, a.v, false, b3.v, (short)0, acc[3], false, false);
        if (k0 + 32 < K) a = an;
    }

    const int rb = half * 8;
#pragma unroll
    for (int j = 0; j < 4; ++j) {
        const int col = n0 + j * 16 + l16;
        const float bc = bias[col];
#pragma unroll
        for (int r = 0; r < 8; ++r) {
            float v = acc[j][r] + bc;
            v = v > 0.f ? v : 0.f;
            size_t idx = (size_t)(m0 + rb + r) * NC + col;
            if constexpr (F32OUT) Cf[idx] = v;
            Cb[idx] = f2bf(v);
        }
    }
}

// ---------------- edge-model GEMM1 with fused gather-concat ----------------
// A row m = concat(x[send[m]], x[recv[m]], e[m])  (K = 384),  NC = 256.
// 16x64 per wave: the random node gathers are amortized over 4 column tiles.

__global__ __launch_bounds__(256, 2) void k_edge_gemm1(
    const unsigned short* __restrict__ xb,
    const unsigned short* __restrict__ eb,
    const int* __restrict__ send,
    const int* __restrict__ recv,
    const unsigned short* __restrict__ Wt,   // (256, 384) bf16
    const float* __restrict__ bias,
    unsigned short* __restrict__ Cb) {

    const int lane = threadIdx.x & 31;
    const int wave = threadIdx.x >> 5;
    const int l16  = lane & 15;
    const int half = lane >> 4;
    const int m0 = (blockIdx.x * 2 + (wave >> 2)) * 16;  // 2 m-tiles per block
    const int m  = m0 + l16;
    const int n0 = (wave & 3) * 64;

    const unsigned short* seg0 = xb + (size_t)send[m] * DN;
    const unsigned short* seg1 = xb + (size_t)recv[m] * DN;
    const unsigned short* seg2 = eb + (size_t)m * DN;
    const unsigned short* Br = Wt + (size_t)(n0 + l16) * 384;

    v8f acc[4];
#pragma unroll
    for (int j = 0; j < 4; ++j) acc[j] = (v8f){0.f,0.f,0.f,0.f,0.f,0.f,0.f,0.f};

    Frag a;
    {
        const int kl = half * 8;
        a.q[0] = *(const v4u*)(seg0 + kl);
        a.q[1] = *(const v4u*)(seg0 + kl + 16);
    }

#pragma unroll
    for (int k0 = 0; k0 < 384; k0 += 32) {
        const int kb = k0 + half * 8;
        Frag b0, b1, b2, b3;
        b0.q[0] = *(const v4u*)(Br + kb);
        b0.q[1] = *(const v4u*)(Br + kb + 16);
        b1.q[0] = *(const v4u*)(Br + (size_t)16 * 384 + kb);
        b1.q[1] = *(const v4u*)(Br + (size_t)16 * 384 + kb + 16);
        b2.q[0] = *(const v4u*)(Br + (size_t)32 * 384 + kb);
        b2.q[1] = *(const v4u*)(Br + (size_t)32 * 384 + kb + 16);
        b3.q[0] = *(const v4u*)(Br + (size_t)48 * 384 + kb);
        b3.q[1] = *(const v4u*)(Br + (size_t)48 * 384 + kb + 16);
        Frag an;
        if (k0 + 32 < 384) {
            const int k1 = k0 + 32;
            const unsigned short* s = (k1 < 128) ? seg0 : (k1 < 256) ? seg1 : seg2;
            const int kl = (k1 & 127) + half * 8;
            an.q[0] = *(const v4u*)(s + kl);
            an.q[1] = *(const v4u*)(s + kl + 16);
        }
        acc[0] = __builtin_amdgcn_wmma_f32_16x16x32_bf16(
            false, a.v, false, b0.v, (short)0, acc[0], false, false);
        acc[1] = __builtin_amdgcn_wmma_f32_16x16x32_bf16(
            false, a.v, false, b1.v, (short)0, acc[1], false, false);
        acc[2] = __builtin_amdgcn_wmma_f32_16x16x32_bf16(
            false, a.v, false, b2.v, (short)0, acc[2], false, false);
        acc[3] = __builtin_amdgcn_wmma_f32_16x16x32_bf16(
            false, a.v, false, b3.v, (short)0, acc[3], false, false);
        if (k0 + 32 < 384) a = an;
    }

    const int rb = half * 8;
#pragma unroll
    for (int j = 0; j < 4; ++j) {
        const int col = n0 + j * 16 + l16;
        const float bc = bias[col];
#pragma unroll
        for (int r = 0; r < 8; ++r) {
            float v = acc[j][r] + bc;
            v = v > 0.f ? v : 0.f;
            Cb[(size_t)(m0 + rb + r) * 256 + col] = f2bf(v);
        }
    }
}

// ---------------- node-model GEMM1 with fused concat[x, agg] ----------------
// A row i = concat(x[i] (bf16), agg[i] (f32 -> bf16))  (K = 256),  NC = 256.

__global__ __launch_bounds__(256, 2) void k_node_gemm1(
    const unsigned short* __restrict__ xb,
    const float* __restrict__ agg,
    const unsigned short* __restrict__ Wt,   // (256, 256) bf16
    const float* __restrict__ bias,
    unsigned short* __restrict__ Cb) {

    const int lane = threadIdx.x & 31;
    const int wave = threadIdx.x >> 5;
    const int l16  = lane & 15;
    const int half = lane >> 4;
    const int m0 = (blockIdx.x * 2 + (wave >> 2)) * 16;
    const int m  = m0 + l16;
    const int n0 = (wave & 3) * 64;

    const unsigned short* Ax = xb  + (size_t)m * DN;
    const float*          Ag = agg + (size_t)m * DN;
    const unsigned short* Br = Wt  + (size_t)(n0 + l16) * 256;

    v8f acc[4];
#pragma unroll
    for (int j = 0; j < 4; ++j) acc[j] = (v8f){0.f,0.f,0.f,0.f,0.f,0.f,0.f,0.f};

#pragma unroll
    for (int k0 = 0; k0 < 256; k0 += 32) {
        const int kb = k0 + half * 8;
        Frag b0, b1, b2, b3;
        b0.q[0] = *(const v4u*)(Br + kb);
        b0.q[1] = *(const v4u*)(Br + kb + 16);
        b1.q[0] = *(const v4u*)(Br + (size_t)16 * 256 + kb);
        b1.q[1] = *(const v4u*)(Br + (size_t)16 * 256 + kb + 16);
        b2.q[0] = *(const v4u*)(Br + (size_t)32 * 256 + kb);
        b2.q[1] = *(const v4u*)(Br + (size_t)32 * 256 + kb + 16);
        b3.q[0] = *(const v4u*)(Br + (size_t)48 * 256 + kb);
        b3.q[1] = *(const v4u*)(Br + (size_t)48 * 256 + kb + 16);
        Frag a;
        if (k0 < 128) {
            const int kl = k0 + half * 8;
            a.q[0] = *(const v4u*)(Ax + kl);
            a.q[1] = *(const v4u*)(Ax + kl + 16);
        } else {
            const int kl = (k0 - 128) + half * 8;
            v4f f0 = *(const v4f*)(Ag + kl);
            v4f f1 = *(const v4f*)(Ag + kl + 4);
            v4f f2 = *(const v4f*)(Ag + kl + 16);
            v4f f3 = *(const v4f*)(Ag + kl + 20);
#pragma unroll
            for (int j = 0; j < 4; ++j) {
                a.h[j]      = f2bf(f0[j]);
                a.h[4 + j]  = f2bf(f1[j]);
                a.h[8 + j]  = f2bf(f2[j]);
                a.h[12 + j] = f2bf(f3[j]);
            }
        }
        acc[0] = __builtin_amdgcn_wmma_f32_16x16x32_bf16(
            false, a.v, false, b0.v, (short)0, acc[0], false, false);
        acc[1] = __builtin_amdgcn_wmma_f32_16x16x32_bf16(
            false, a.v, false, b1.v, (short)0, acc[1], false, false);
        acc[2] = __builtin_amdgcn_wmma_f32_16x16x32_bf16(
            false, a.v, false, b2.v, (short)0, acc[2], false, false);
        acc[3] = __builtin_amdgcn_wmma_f32_16x16x32_bf16(
            false, a.v, false, b3.v, (short)0, acc[3], false, false);
    }

    const int rb = half * 8;
#pragma unroll
    for (int j = 0; j < 4; ++j) {
        const int col = n0 + j * 16 + l16;
        const float bc = bias[col];
#pragma unroll
        for (int r = 0; r < 8; ++r) {
            float v = acc[j][r] + bc;
            v = v > 0.f ? v : 0.f;
            Cb[(size_t)(m0 + rb + r) * 256 + col] = f2bf(v);
        }
    }
}

// ---------------- scatter-add (segment_sum) ----------------

__global__ void k_segment_sum(const unsigned short* __restrict__ eb,
                              const int* __restrict__ recv,
                              float* __restrict__ agg) {
    int i = blockIdx.x * blockDim.x + threadIdx.x;
    if (i >= EE * DN) return;
    int j = i >> 7;          // edge
    int d = i & (DN - 1);    // feature
    atomicAdd(&agg[(size_t)recv[j] * DN + d], bf2f(eb[i]));
}

// ---------------- per-node row sum (global_add_pool per node) ----------------

__global__ __launch_bounds__(256) void k_row_sum(const float* __restrict__ x,
                                                 float* __restrict__ out, int n) {
    const int lane = threadIdx.x & 31;
    const int row  = blockIdx.x * 8 + (threadIdx.x >> 5);
    if (row >= n) return;
    const float* p = x + (size_t)row * DN;
    float s = p[lane] + p[lane + 32] + p[lane + 64] + p[lane + 96];
#pragma unroll
    for (int o = 16; o; o >>= 1) s += __shfl_xor(s, o, 32);
    if (lane == 0) out[row] = s;
}

// ---------------- host-side pipeline ----------------

extern "C" void kernel_launch(void* const* d_in, const int* in_sizes, int n_in,
                              void* d_out, int out_size, void* d_ws, size_t ws_size,
                              hipStream_t stream) {
    const float* nodes     = (const float*)d_in[0];
    const float* edges     = (const float*)d_in[1];
    const int*   senders   = (const int*)d_in[2];
    const int*   receivers = (const int*)d_in[3];
    const float* Wn1 = (const float*)d_in[4];  const float* bn1 = (const float*)d_in[5];
    const float* Wn2 = (const float*)d_in[6];  const float* bn2 = (const float*)d_in[7];
    const float* We1 = (const float*)d_in[8];  const float* be1 = (const float*)d_in[9];
    const float* We2 = (const float*)d_in[10]; const float* be2 = (const float*)d_in[11];
    const float* Wed1 = (const float*)d_in[12]; const float* bed1 = (const float*)d_in[13];
    const float* Wed2 = (const float*)d_in[14]; const float* bed2 = (const float*)d_in[15];
    const float* Wnd1 = (const float*)d_in[16]; const float* bnd1 = (const float*)d_in[17];
    const float* Wnd2 = (const float*)d_in[18]; const float* bnd2 = (const float*)d_in[19];
    float* out = (float*)d_out;

    // workspace carve-up (256B aligned)
    char* ws = (char*)d_ws;
    size_t off = 0;
    auto alloc = [&](size_t bytes) -> char* {
        off = (off + 255) & ~(size_t)255;
        char* p = ws + off;
        off += bytes;
        return p;
    };
    unsigned short* e_bf  = (unsigned short*)alloc((size_t)EE * DN * 2); // also edges_bf
    unsigned short* hid_e = (unsigned short*)alloc((size_t)EE * HN * 2);
    unsigned short* x_bf  = (unsigned short*)alloc((size_t)NN * DN * 2); // also nodes_bf
    float*          x_f32 = (float*)alloc((size_t)NN * DN * 4);
    float*          agg   = (float*)alloc((size_t)NN * DN * 4);
    unsigned short* hid_n = (unsigned short*)alloc((size_t)NN * HN * 2);
    unsigned short* Wn1t  = (unsigned short*)alloc((size_t)DN * HN * 2);
    unsigned short* Wn2t  = (unsigned short*)alloc((size_t)HN * DN * 2);
    unsigned short* We1t  = (unsigned short*)alloc((size_t)DN * HN * 2);
    unsigned short* We2t  = (unsigned short*)alloc((size_t)HN * DN * 2);
    unsigned short* Wed1t = (unsigned short*)alloc((size_t)3 * DN * HN * 2);
    unsigned short* Wed2t = (unsigned short*)alloc((size_t)HN * DN * 2);
    unsigned short* Wnd1t = (unsigned short*)alloc((size_t)2 * DN * HN * 2);
    unsigned short* Wnd2t = (unsigned short*)alloc((size_t)HN * DN * 2);

    const int T = 256;
    // input conversion
    k_cvt_bf16<<<(NN * DN + T - 1) / T, T, 0, stream>>>(nodes, x_bf, NN * DN);
    k_cvt_bf16<<<(EE * DN + T - 1) / T, T, 0, stream>>>(edges, e_bf, EE * DN);
    // weight transposes (f32 -> bf16, (K,NC) -> (NC,K))
    k_transpose_w<<<(DN * HN + T - 1) / T, T, 0, stream>>>(Wn1, Wn1t, DN, HN);
    k_transpose_w<<<(HN * DN + T - 1) / T, T, 0, stream>>>(Wn2, Wn2t, HN, DN);
    k_transpose_w<<<(DN * HN + T - 1) / T, T, 0, stream>>>(We1, We1t, DN, HN);
    k_transpose_w<<<(HN * DN + T - 1) / T, T, 0, stream>>>(We2, We2t, HN, DN);
    k_transpose_w<<<(3 * DN * HN + T - 1) / T, T, 0, stream>>>(Wed1, Wed1t, 3 * DN, HN);
    k_transpose_w<<<(HN * DN + T - 1) / T, T, 0, stream>>>(Wed2, Wed2t, HN, DN);
    k_transpose_w<<<(2 * DN * HN + T - 1) / T, T, 0, stream>>>(Wnd1, Wnd1t, 2 * DN, HN);
    k_transpose_w<<<(HN * DN + T - 1) / T, T, 0, stream>>>(Wnd2, Wnd2t, HN, DN);

    // grid sizes: m-tiles / m-tiles-per-block
    const int gn256 = (NN / 16 + 1) / 2;        // NC=256: MTB=2 -> 625
    const int gn128 = (NN / 16 + 3) / 4;        // NC=128: MTB=4 -> 313 (guarded)
    const int ge256 = (EE / 16) / 2;            // 10000
    const int ge128 = (EE / 16) / 4;            // 5000

    // embed node MLP: x = relu(relu(nodes@Wn1+bn1)@Wn2+bn2)
    k_gemm_bias_relu<DN, HN, false><<<gn256, T, 0, stream>>>(x_bf, Wn1t, bn1, nullptr, hid_n, NN);
    k_gemm_bias_relu<HN, DN, true ><<<gn128, T, 0, stream>>>(hid_n, Wn2t, bn2, x_f32, x_bf, NN);
    // embed edge MLP: e = relu(relu(edges@We1+be1)@We2+be2)
    k_gemm_bias_relu<DN, HN, false><<<ge256, T, 0, stream>>>(e_bf, We1t, be1, nullptr, hid_e, EE);
    k_gemm_bias_relu<HN, DN, false><<<ge128, T, 0, stream>>>(hid_e, We2t, be2, nullptr, e_bf, EE);

    for (int it = 0; it < 2; ++it) {
        // edge model
        k_edge_gemm1<<<ge256, T, 0, stream>>>(x_bf, e_bf, senders, receivers, Wed1t, bed1, hid_e);
        k_gemm_bias_relu<HN, DN, false><<<ge128, T, 0, stream>>>(hid_e, Wed2t, bed2, nullptr, e_bf, EE);
        // aggregate to receivers
        k_zero_f32<<<(NN * DN + T - 1) / T, T, 0, stream>>>(agg, NN * DN);
        k_segment_sum<<<(EE * DN + T - 1) / T, T, 0, stream>>>(e_bf, receivers, agg);
        // node model
        k_node_gemm1<<<gn256, T, 0, stream>>>(x_bf, agg, Wnd1t, bnd1, hid_n);
        k_gemm_bias_relu<HN, DN, true><<<gn128, T, 0, stream>>>(hid_n, Wnd2t, bnd2, x_f32, x_bf, NN);
        // per-node pooled sum
        k_row_sum<<<(NN + 7) / 8, T, 0, stream>>>(x_f32, out + (size_t)it * NN, NN);
    }
    (void)in_sizes; (void)n_in; (void)out_size; (void)ws_size;
}
